// self_attention_16638703305389
// MI455X (gfx1250) — matrix operand order repaired and checked by reading, not compiled, not run on hardware
//
#include <hip/hip_runtime.h>
#include <hip/hip_fp16.h>

// MI455X / gfx1250 (CDNA5, wave32). Compute-bound attention: use
// v_wmma_f32_16x16x32_f16 (f16 in, f32 accumulate) for both GEMMs.

typedef __attribute__((ext_vector_type(16))) _Float16 v16h;
typedef __attribute__((ext_vector_type(8)))  float    v8f;

#define BATCH 4
#define CCH   256
#define NPIX  4096      // H*W = 64*64
#define DQK   32
#define BN_EPS 1e-5f

static __device__ __forceinline__ v8f wmma_f16(v16h a, v16h b, v8f c) {
  return __builtin_amdgcn_wmma_f32_16x16x32_f16(false, a, false, b, (short)0, c,
                                                false, false);
}

// A-matrix fragment (16x32 f16): lane holds row m = lane&15.
// element e: K = hi*16 + e   (hi = lane>>4)
// => one contiguous 16-half (32B) run at offset hi*16.
static __device__ __forceinline__ v16h load_a_frag(const _Float16* rowptr, int hi) {
  v16h f;
  reinterpret_cast<uint4*>(&f)[0] = *reinterpret_cast<const uint4*>(rowptr + hi * 16);
  reinterpret_cast<uint4*>(&f)[1] = *reinterpret_cast<const uint4*>(rowptr + hi * 16 + 8);
  return f;
}

// B-matrix fragment (32x16 f16): lane holds column n = lane&15.
// element e: K = hi*16 + e  => one contiguous 16-half (32B) run.
static __device__ __forceinline__ v16h load_b_frag(const _Float16* colptr, int hi) {
  v16h f;
  const uint4* p = reinterpret_cast<const uint4*>(colptr + hi * 16);
  reinterpret_cast<uint4*>(&f)[0] = p[0];
  reinterpret_cast<uint4*>(&f)[1] = p[1];
  return f;
}

// -------- Kernel 1: fused q/k/v 1x1-conv projections (f32 -> f16) -----------
// qT: [B][N][32], kT: [B][N][32], v: [B][C][N]
__global__ __launch_bounds__(256) void proj_kernel(
    const float* __restrict__ x,
    const float* __restrict__ Wq, const float* __restrict__ bq,
    const float* __restrict__ Wk, const float* __restrict__ bk,
    const float* __restrict__ Wv, const float* __restrict__ bv,
    _Float16* __restrict__ qT, _Float16* __restrict__ kT,
    _Float16* __restrict__ v) {
  __shared__ float xs[CCH * 32];
  const int b = blockIdx.x;
  const int pix0 = blockIdx.y * 32;
  const float* xb = x + (size_t)b * CCH * NPIX + pix0;
  for (int idx = threadIdx.x; idx < CCH * 32; idx += 256) {
    int c = idx >> 5, p = idx & 31;
    xs[idx] = xb[(size_t)c * NPIX + p];
  }
  __syncthreads();
  const int p = threadIdx.x & 31;        // pixel within tile (lane id)
  const int og = threadIdx.x >> 5;       // wave-uniform output-channel group
  const int pix = pix0 + p;
  for (int o = og; o < CCH + 2 * DQK; o += 8) {
    const float* Wrow;
    float acc;
    if (o < DQK)            { Wrow = Wq + o * CCH;            acc = bq[o]; }
    else if (o < 2 * DQK)   { Wrow = Wk + (o - DQK) * CCH;    acc = bk[o - DQK]; }
    else                    { Wrow = Wv + (o - 2*DQK) * CCH;  acc = bv[o - 2*DQK]; }
#pragma unroll 8
    for (int c = 0; c < CCH; ++c) acc += Wrow[c] * xs[c * 32 + p];
    _Float16 h = (_Float16)acc;
    if (o < DQK)          qT[((size_t)b * NPIX + pix) * DQK + o] = h;
    else if (o < 2*DQK)   kT[((size_t)b * NPIX + pix) * DQK + (o - DQK)] = h;
    else                  v[((size_t)b * CCH + (o - 2*DQK)) * NPIX + pix] = h;
  }
}

// -------- Kernel 2: per-row softmax stats (max_i, 1/Z_i) via WMMA ----------
// One wave per (batch, 16-row i-tile). j-loop unrolled x4: 4 independent
// WMMAs per iteration (hides WMMA->VALU hazards + batches loads into one
// clause), merged online-LSE update = 5 exps per 4 logits.
__global__ __launch_bounds__(32) void stats_kernel(
    const _Float16* __restrict__ qT, const _Float16* __restrict__ kT,
    float* __restrict__ rowmax, float* __restrict__ rowinvz) {
  const int b = blockIdx.x;
  const int i0 = blockIdx.y * 16;
  const int lane = threadIdx.x;
  const int n = lane & 15, hi = lane >> 4;
  const v16h af = load_a_frag(qT + ((size_t)b * NPIX + i0 + n) * DQK, hi);
  const _Float16* kbase = kT + ((size_t)b * NPIX + n) * DQK;
  float m8[8], s8[8];
#pragma unroll
  for (int r = 0; r < 8; ++r) { m8[r] = -1e30f; s8[r] = 0.f; }
  for (int jt = 0; jt < NPIX / 16; jt += 4) {
    v16h bf0 = load_b_frag(kbase + (size_t)(jt + 0) * 16 * DQK, hi);
    v16h bf1 = load_b_frag(kbase + (size_t)(jt + 1) * 16 * DQK, hi);
    v16h bf2 = load_b_frag(kbase + (size_t)(jt + 2) * 16 * DQK, hi);
    v16h bf3 = load_b_frag(kbase + (size_t)(jt + 3) * 16 * DQK, hi);
    v8f z = {};
    v8f d0 = wmma_f16(af, bf0, z);
    v8f d1 = wmma_f16(af, bf1, z);
    v8f d2 = wmma_f16(af, bf2, z);
    v8f d3 = wmma_f16(af, bf3, z);
#pragma unroll
    for (int r = 0; r < 8; ++r) {
      float xm = fmaxf(fmaxf(d0[r], d1[r]), fmaxf(d2[r], d3[r]));
      float nm = fmaxf(m8[r], xm);
      s8[r] = s8[r] * __expf(m8[r] - nm) +
              (__expf(d0[r] - nm) + __expf(d1[r] - nm)) +
              (__expf(d2[r] - nm) + __expf(d3[r] - nm));
      m8[r] = nm;
    }
  }
#pragma unroll
  for (int off = 1; off < 16; off <<= 1) {   // stays within each half-wave
#pragma unroll
    for (int r = 0; r < 8; ++r) {
      float om = __shfl_xor(m8[r], off, 32);
      float os = __shfl_xor(s8[r], off, 32);
      float nm = fmaxf(m8[r], om);
      s8[r] = s8[r] * __expf(m8[r] - nm) + os * __expf(om - nm);
      m8[r] = nm;
    }
  }
  if (n == 0) {
#pragma unroll
    for (int r = 0; r < 8; ++r) {
      int i = i0 + hi * 8 + r;   // D-layout: lanes16-31 carry rows 8..15
      rowmax[(size_t)b * NPIX + i] = m8[r];
      rowinvz[(size_t)b * NPIX + i] = 1.0f / s8[r];
    }
  }
}

// -------- Kernel 3: o[c,j] = sum_i v[c,i] * exp(l[i,j]-m_i)/Z_i ------------
// One wave per (batch, 16-col j-tile, 128-channel half). K-loop over i in
// chunks of 32: 2 logits WMMA -> exp/scale -> half-wave shuffle into a
// B-fragment -> 8 output WMMAs. Q-fragments prefetched one chunk ahead,
// V-fragments pipelined one tile ahead (all branch-free: EXEC stays full).
__global__ __launch_bounds__(32) void attn_out_kernel(
    const _Float16* __restrict__ qT, const _Float16* __restrict__ kT,
    const _Float16* __restrict__ v,
    const float* __restrict__ rowmax, const float* __restrict__ rowinvz,
    float* __restrict__ o) {
  const int b = blockIdx.x;
  const int j0 = blockIdx.y * 16;
  const int chalf = blockIdx.z;             // 0/1 -> channels [0,128)/[128,256)
  const int lane = threadIdx.x;
  const int n = lane & 15, hi = lane >> 4;
  const v16h bk = load_b_frag(kT + ((size_t)b * NPIX + j0 + n) * DQK, hi);
  const float* mb = rowmax + (size_t)b * NPIX;
  const float* zb = rowinvz + (size_t)b * NPIX;
  const _Float16* qbase = qT + ((size_t)b * NPIX + n) * DQK;
  const _Float16* vbase = v + ((size_t)b * CCH + chalf * 128 + n) * NPIX;
  v8f acc[8];
#pragma unroll
  for (int t = 0; t < 8; ++t) acc[t] = (v8f){};

  v16h aq0 = load_a_frag(qbase, hi);
  v16h aq1 = load_a_frag(qbase + (size_t)16 * DQK, hi);
  const int NCHUNK = NPIX / 32;
  for (int ic = 0; ic < NCHUNK; ++ic) {
    const int i0 = ic * 32;
    // Prefetch next chunk's Q fragments (clamped -> no divergence).
    const int inext = (ic + 1 < NCHUNK) ? (i0 + 32) : i0;
    v16h naq0 = load_a_frag(qbase + (size_t)inext * DQK, hi);
    v16h naq1 = load_a_frag(qbase + (size_t)(inext + 16) * DQK, hi);
    v8f z = {};
    v8f d0 = wmma_f16(aq0, bk, z);
    v8f d1 = wmma_f16(aq1, bk, z);
    // Row stats for the 32 i's of this chunk (16B-aligned vector loads).
    float mv0[8], zv0[8], mv1[8], zv1[8];
    *(float4*)&mv0[0] = *(const float4*)(mb + i0 + hi * 8);
    *(float4*)&mv0[4] = *(const float4*)(mb + i0 + hi * 8 + 4);
    *(float4*)&zv0[0] = *(const float4*)(zb + i0 + hi * 8);
    *(float4*)&zv0[4] = *(const float4*)(zb + i0 + hi * 8 + 4);
    *(float4*)&mv1[0] = *(const float4*)(mb + i0 + 16 + hi * 8);
    *(float4*)&mv1[4] = *(const float4*)(mb + i0 + 16 + hi * 8 + 4);
    *(float4*)&zv1[0] = *(const float4*)(zb + i0 + 16 + hi * 8);
    *(float4*)&zv1[4] = *(const float4*)(zb + i0 + 16 + hi * 8 + 4);
    float a0[8], a1[8];
#pragma unroll
    for (int r = 0; r < 8; ++r) {
      a0[r] = __expf(d0[r] - mv0[r]) * zv0[r];
      a1[r] = __expf(d1[r] - mv1[r]) * zv1[r];
    }
    // D layout (lane=column, rows split across half-waves) -> B layout
    // (lane=column, K = hi*16+e): swap row-halves via shfl_xor(16).
    v16h wf;
#pragma unroll
    for (int r = 0; r < 8; ++r) {
      float t0 = __shfl_xor(a0[r], 16, 32);
      float t1 = __shfl_xor(a1[r], 16, 32);
      wf[r]     = (_Float16)(hi ? t1 : a0[r]);   // K = hi*16 + r
      wf[8 + r] = (_Float16)(hi ? a1[r] : t0);   // K = hi*16 + 8 + r
    }
    // 8 output WMMAs, V-fragment loads pipelined one tile ahead.
    v16h vf = load_a_frag(vbase + i0, hi);
#pragma unroll
    for (int t = 0; t < 8; ++t) {
      const int tn = (t + 1 < 8) ? (t + 1) : t;
      v16h vfn = load_a_frag(vbase + (size_t)tn * 16 * NPIX + i0, hi);
      acc[t] = wmma_f16(vf, wf, acc[t]);
      vf = vfn;
    }
    aq0 = naq0;
    aq1 = naq1;
  }
#pragma unroll
  for (int t = 0; t < 8; ++t) {
    const int cc = chalf * 128 + t * 16 + hi * 8;
#pragma unroll
    for (int r = 0; r < 8; ++r)
      o[((size_t)b * CCH + cc + r) * NPIX + j0 + n] = acc[t][r];
  }
}

// -------- Kernel 4: BatchNorm batch stats -> per-channel affine ------------
__global__ __launch_bounds__(256) void bn_stats_kernel(
    const float* __restrict__ o, const float* __restrict__ gamma,
    const float* __restrict__ beta, float* __restrict__ scale,
    float* __restrict__ shift) {
  const int c = blockIdx.x;
  float s = 0.f, s2 = 0.f;
  for (int idx = threadIdx.x; idx < BATCH * NPIX; idx += 256) {
    int b = idx >> 12, n = idx & (NPIX - 1);
    float val = o[((size_t)b * CCH + c) * NPIX + n];
    s += val;
    s2 += val * val;
  }
#pragma unroll
  for (int off = 16; off; off >>= 1) {
    s += __shfl_xor(s, off, 32);
    s2 += __shfl_xor(s2, off, 32);
  }
  __shared__ float ss[8], ss2[8];
  if ((threadIdx.x & 31) == 0) { ss[threadIdx.x >> 5] = s; ss2[threadIdx.x >> 5] = s2; }
  __syncthreads();
  if (threadIdx.x == 0) {
    float ts = 0.f, ts2 = 0.f;
    for (int w = 0; w < 8; ++w) { ts += ss[w]; ts2 += ss2[w]; }
    const float invn = 1.0f / (BATCH * NPIX);
    float mean = ts * invn;
    float var = ts2 * invn - mean * mean;
    float g = gamma[c] * rsqrtf(var + BN_EPS);
    scale[c] = g * 1e-5f;
    shift[c] = (beta[c] - mean * g) * 1e-5f;
  }
}

// -------- Kernel 5: residual + affine --------------------------------------
__global__ __launch_bounds__(256) void final_kernel(
    const float* __restrict__ x, const float* __restrict__ o,
    const float* __restrict__ scale, const float* __restrict__ shift,
    float* __restrict__ out) {
  size_t idx = (size_t)blockIdx.x * 256 + threadIdx.x;
  int c = (int)((idx >> 12) & (CCH - 1));   // idx / NPIX % C
  out[idx] = x[idx] + o[idx] * scale[c] + shift[c];
}

extern "C" void kernel_launch(void* const* d_in, const int* in_sizes, int n_in,
                              void* d_out, int out_size, void* d_ws, size_t ws_size,
                              hipStream_t stream) {
  const float* x  = (const float*)d_in[0];
  const float* Wq = (const float*)d_in[1];
  const float* bq = (const float*)d_in[2];
  const float* Wk = (const float*)d_in[3];
  const float* bk = (const float*)d_in[4];
  const float* Wv = (const float*)d_in[5];
  const float* bv = (const float*)d_in[6];
  const float* gamma = (const float*)d_in[7];
  const float* beta  = (const float*)d_in[8];
  float* out = (float*)d_out;

  // Workspace carve-up (all offsets 256B-aligned)
  char* ws = (char*)d_ws;
  size_t off = 0;
  _Float16* qT = (_Float16*)(ws + off); off += (size_t)BATCH * NPIX * DQK * 2;   // 1 MB
  _Float16* kT = (_Float16*)(ws + off); off += (size_t)BATCH * NPIX * DQK * 2;   // 1 MB
  _Float16* vp = (_Float16*)(ws + off); off += (size_t)BATCH * CCH * NPIX * 2;   // 8 MB
  float* rowmax  = (float*)(ws + off); off += (size_t)BATCH * NPIX * 4;          // 64 KB
  float* rowinvz = (float*)(ws + off); off += (size_t)BATCH * NPIX * 4;          // 64 KB
  float* obuf    = (float*)(ws + off); off += (size_t)BATCH * CCH * NPIX * 4;    // 16 MB
  float* scale   = (float*)(ws + off); off += 256 * 4;
  float* shift   = (float*)(ws + off); off += 256 * 4;

  proj_kernel<<<dim3(BATCH, NPIX / 32), 256, 0, stream>>>(
      x, Wq, bq, Wk, bk, Wv, bv, qT, kT, vp);
  stats_kernel<<<dim3(BATCH, NPIX / 16), 32, 0, stream>>>(
      qT, kT, rowmax, rowinvz);
  attn_out_kernel<<<dim3(BATCH, NPIX / 16, 2), 32, 0, stream>>>(
      qT, kT, vp, rowmax, rowinvz, obuf);
  bn_stats_kernel<<<dim3(CCH), 256, 0, stream>>>(obuf, gamma, beta, scale, shift);
  final_kernel<<<dim3((BATCH * CCH * NPIX) / 256), 256, 0, stream>>>(
      x, obuf, scale, shift, out);
}
